// GNN_64725157151112
// MI455X (gfx1250) — matrix operand order repaired and checked by reading, not compiled, not run on hardware
//
#include <hip/hip_runtime.h>
#include <hip/hip_bf16.h>
#include <math.h>

// ---------------------------------------------------------------------------
// Types for CDNA5 WMMA (wave32): 16x16x32 bf16 -> f32 accumulate
// ---------------------------------------------------------------------------
typedef __bf16 v16bf __attribute__((ext_vector_type(16)));
typedef __bf16 v8bf  __attribute__((ext_vector_type(8)));
typedef float  v8f   __attribute__((ext_vector_type(8)));

#define IN_DIM 256
#define HID    128
#define LEAKY  0.01f
#define MROWS  64      // rows per block (4 M-tiles of 16)

// ---------------------------------------------------------------------------
// Utility: fill a float buffer with a constant
// ---------------------------------------------------------------------------
__global__ __launch_bounds__(256) void k_fill(float* __restrict__ p, float v, int n) {
    int i = blockIdx.x * 256 + threadIdx.x;
    if (i < n) p[i] = v;
}

// ---------------------------------------------------------------------------
// WMMA GEMM + bias:  Z[M,128] = A[M,K] @ W[128,K]^T + b[128]
// Block = 256 threads = 8 waves; block computes 64 rows x 128 cols.
// Wave w owns N-tile [16w,16w+16) and 4 M-tiles; one B fragment feeds 4 wmmas.
// A K-slab (64x32) is staged in LDS as bf16 (converted at global-load time).
//
// Fragment layouts used here (wave32, 16-bit elements; A and B share one K order):
//   A 16x32:  lane l -> row l&15; K = 16*(l>>4) .. +16   (contiguous)
//   B 32x16:  lane l -> col l&15; K = 16*(l>>4) .. +16   (contiguous)
//   C/D 16x16: lane l, VGPR r -> M = r + 8*(l>>4), N = l&15
// ---------------------------------------------------------------------------
template <int K_DIM>
__global__ __launch_bounds__(256) void k_gemm_bias(
    const float* __restrict__ A,     // [M, K_DIM]
    const float* __restrict__ W,     // [128, K_DIM]
    const float* __restrict__ bias,  // [128]
    float* __restrict__ Z,           // [M, 128]
    int M)
{
    __shared__ __bf16 As[MROWS * 32];   // 4 KB

    const int tid     = threadIdx.x;
    const int wave    = tid >> 5;        // 0..7
    const int lane    = tid & 31;
    const int halfSel = lane >> 4;       // 0 or 1
    const int l15     = lane & 15;
    const int mbase   = blockIdx.x * MROWS;
    const int n0      = wave * 16;
    const int kb      = halfSel * 8;     // unused; A K base is 16*halfSel

    v8f c[4] = {{}, {}, {}, {}};

    for (int k0 = 0; k0 < K_DIM; k0 += 32) {
        __syncthreads();  // protect As reuse across iterations
        {   // stage 64x32 f32 -> bf16 slab: 2048 elems, 8 per thread
            const int idx = tid * 8;
            const int r   = idx >> 5;                    // row 0..63
            const int col = idx & 31;                    // 0,8,16,24
            const int rg  = min(mbase + r, M - 1);       // branch-free clamp
            const float* ap = &A[(size_t)rg * K_DIM + k0 + col];
            const float4 v0 = *(const float4*)(ap);
            const float4 v1 = *(const float4*)(ap + 4);
            __builtin_prefetch(ap + 32, 0, 3);           // speculative: safe OOB
            v8bf t;
            t[0] = (__bf16)v0.x; t[1] = (__bf16)v0.y;
            t[2] = (__bf16)v0.z; t[3] = (__bf16)v0.w;
            t[4] = (__bf16)v1.x; t[5] = (__bf16)v1.y;
            t[6] = (__bf16)v1.z; t[7] = (__bf16)v1.w;
            *(v8bf*)&As[r * 32 + col] = t;               // one ds_store_b128
        }

        // B fragment: lane holds column d = n0+l15 of B == row d of W;
        // contiguous K run of 16 starting at 16*halfSel (same K order as A).
        // Issued before the barrier so W's global latency overlaps it.
        v16bf bfrag;
        {
            const float* wr = &W[(size_t)(n0 + l15) * K_DIM + k0 + halfSel * 16];
            const float4 w0 = *(const float4*)(wr);
            const float4 w1 = *(const float4*)(wr + 4);
            const float4 w2 = *(const float4*)(wr + 8);
            const float4 w3 = *(const float4*)(wr + 12);
            bfrag[0]  = (__bf16)w0.x; bfrag[1]  = (__bf16)w0.y;
            bfrag[2]  = (__bf16)w0.z; bfrag[3]  = (__bf16)w0.w;
            bfrag[4]  = (__bf16)w1.x; bfrag[5]  = (__bf16)w1.y;
            bfrag[6]  = (__bf16)w1.z; bfrag[7]  = (__bf16)w1.w;
            bfrag[8]  = (__bf16)w2.x; bfrag[9]  = (__bf16)w2.y;
            bfrag[10] = (__bf16)w2.z; bfrag[11] = (__bf16)w2.w;
            bfrag[12] = (__bf16)w3.x; bfrag[13] = (__bf16)w3.y;
            bfrag[14] = (__bf16)w3.z; bfrag[15] = (__bf16)w3.w;
        }

        __syncthreads();

        #pragma unroll
        for (int t = 0; t < 4; ++t) {
            // A fragment for M-tile t: two 16B LDS loads (contiguous K run of 16 at 16*halfSel)
            const __bf16* row = &As[(t * 16 + l15) * 32];
            const v8bf lo = *(const v8bf*)(row + halfSel * 16);
            const v8bf hi = *(const v8bf*)(row + halfSel * 16 + 8);
            const v16bf afrag = __builtin_shufflevector(
                lo, hi, 0, 1, 2, 3, 4, 5, 6, 7, 8, 9, 10, 11, 12, 13, 14, 15);
            c[t] = __builtin_amdgcn_wmma_f32_16x16x32_bf16(
                       false, afrag, false, bfrag, (short)0, c[t], false, false);
        }
    }

    // Bias + store. Lanes 0..15 write contiguous columns of one row -> coalesced.
    const float bv = bias[n0 + l15];
    if (mbase + MROWS <= M) {            // uniform (scalar) branch: full block
        #pragma unroll
        for (int t = 0; t < 4; ++t)
            #pragma unroll
            for (int r = 0; r < 8; ++r) {
                const int m = mbase + t * 16 + r + 8 * halfSel;
                Z[(size_t)m * HID + n0 + l15] = c[t][r] + bv;
            }
    } else {                             // ragged last block only
        #pragma unroll
        for (int t = 0; t < 4; ++t)
            #pragma unroll
            for (int r = 0; r < 8; ++r) {
                const int m = mbase + t * 16 + r + 8 * halfSel;
                if (m < M) Z[(size_t)m * HID + n0 + l15] = c[t][r] + bv;
            }
    }
}

// ---------------------------------------------------------------------------
// Per-node attention scalars: s_src[n] = z[n,:]·a[:128], s_dst[n] = z[n,:]·a[128:]
// One wave per node; lane covers 4 dims; butterfly reduce in wave32.
// ---------------------------------------------------------------------------
__global__ __launch_bounds__(256) void k_node_scores(
    const float* __restrict__ Z, const float* __restrict__ a,
    float* __restrict__ s_src, float* __restrict__ s_dst, int N)
{
    const int node = (blockIdx.x * 256 + threadIdx.x) >> 5;
    const int lane = threadIdx.x & 31;
    if (node >= N) return;

    const float4 z4 = *(const float4*)&Z[(size_t)node * HID + lane * 4];
    const float4 al = *(const float4*)&a[lane * 4];
    const float4 ah = *(const float4*)&a[HID + lane * 4];

    float ss = z4.x * al.x + z4.y * al.y + z4.z * al.z + z4.w * al.w;
    float sd = z4.x * ah.x + z4.y * ah.y + z4.z * ah.z + z4.w * ah.w;
    #pragma unroll
    for (int off = 16; off > 0; off >>= 1) {
        ss += __shfl_xor(ss, off, 32);
        sd += __shfl_xor(sd, off, 32);
    }
    if (lane == 0) { s_src[node] = ss; s_dst[node] = sd; }
}

// ---------------------------------------------------------------------------
// Edge logits + segment max (float atomic max via sign-split int atomics)
// ---------------------------------------------------------------------------
__device__ inline void atomic_max_float(float* addr, float v) {
    if (v >= 0.0f) atomicMax((int*)addr, __float_as_int(v));
    else           atomicMin((unsigned int*)addr, __float_as_uint(v));
}

__global__ __launch_bounds__(256) void k_edge_logits_max(
    const int* __restrict__ src, const int* __restrict__ dst,
    const float* __restrict__ s_src, const float* __restrict__ s_dst,
    const float* __restrict__ abp,
    float* __restrict__ e, float* __restrict__ node_max, int E)
{
    const int i = blockIdx.x * 256 + threadIdx.x;
    if (i >= E) return;
    float v = s_src[src[i]] + s_dst[dst[i]] + abp[0];
    v = (v > 0.0f) ? v : LEAKY * v;   // leaky relu
    e[i] = v;
    atomic_max_float(&node_max[dst[i]], v);
}

// ---------------------------------------------------------------------------
// Edge exp + segment sum (overwrites e with ex)
// ---------------------------------------------------------------------------
__global__ __launch_bounds__(256) void k_edge_exp_sum(
    const int* __restrict__ dst, const float* __restrict__ node_max,
    float* __restrict__ e, float* __restrict__ node_sum, int E)
{
    const int i = blockIdx.x * 256 + threadIdx.x;
    if (i >= E) return;
    const int d = dst[i];
    const float ex = expf(e[i] - node_max[d]);
    e[i] = ex;
    atomicAdd(&node_sum[d], ex);
}

// ---------------------------------------------------------------------------
// Weighted scatter: out[dst,:] += (ex/denom[dst]) * Z[src,:]
// One wave per edge, lane covers 4 dims (float4 gather, 4 f32 atomic adds)
// ---------------------------------------------------------------------------
__global__ __launch_bounds__(256) void k_scatter(
    const int* __restrict__ src, const int* __restrict__ dst,
    const float* __restrict__ ex, const float* __restrict__ node_sum,
    const float* __restrict__ Z, float* __restrict__ out, int E)
{
    const int edge = (blockIdx.x * 256 + threadIdx.x) >> 5;
    const int lane = threadIdx.x & 31;
    if (edge >= E) return;

    const int s = src[edge];
    const int d = dst[edge];
    const float alpha = ex[edge] / node_sum[d];

    const float4 z4 = *(const float4*)&Z[(size_t)s * HID + lane * 4];
    float* o = &out[(size_t)d * HID + lane * 4];
    atomicAdd(o + 0, alpha * z4.x);
    atomicAdd(o + 1, alpha * z4.y);
    atomicAdd(o + 2, alpha * z4.z);
    atomicAdd(o + 3, alpha * z4.w);
}

// ---------------------------------------------------------------------------
// Elementwise ELU in place (alpha=1): x>0 ? x : expm1(x)
// ---------------------------------------------------------------------------
__global__ __launch_bounds__(256) void k_elu(float* __restrict__ p, int n) {
    const int i = blockIdx.x * 256 + threadIdx.x;
    if (i < n) {
        const float x = p[i];
        p[i] = (x > 0.0f) ? x : expm1f(x);
    }
}

// ---------------------------------------------------------------------------
// Host-side layer driver (records kernels onto the stream; graph-capture safe)
// ---------------------------------------------------------------------------
static void run_gat_layer(const float* A, int K_dim,
                          const int* src, const int* dst,
                          const float* W, const float* b, const float* a,
                          const float* ab,
                          float* z, float* s_src, float* s_dst,
                          float* node_max, float* node_sum, float* ebuf,
                          float* accum_out,   // [N,128], will hold elu(layer out)
                          int N, int E, hipStream_t stream)
{
    const int nBlocksN   = (N + 255) / 256;
    const int nBlocksNF  = (N * HID + 255) / 256;
    const int nBlocksE   = (E + 255) / 256;
    const int nBlocksEW  = (E * 32 + 255) / 256;   // one wave per edge
    const int nBlocksM   = (N + MROWS - 1) / MROWS;

    k_fill<<<nBlocksN, 256, 0, stream>>>(node_max, -INFINITY, N);
    k_fill<<<nBlocksN, 256, 0, stream>>>(node_sum, 0.0f, N);
    k_fill<<<nBlocksNF, 256, 0, stream>>>(accum_out, 0.0f, N * HID);

    if (K_dim == IN_DIM)
        k_gemm_bias<IN_DIM><<<nBlocksM, 256, 0, stream>>>(A, W, b, z, N);
    else
        k_gemm_bias<HID><<<nBlocksM, 256, 0, stream>>>(A, W, b, z, N);

    k_node_scores<<<(N * 32 + 255) / 256, 256, 0, stream>>>(z, a, s_src, s_dst, N);
    k_edge_logits_max<<<nBlocksE, 256, 0, stream>>>(src, dst, s_src, s_dst, ab,
                                                    ebuf, node_max, E);
    k_edge_exp_sum<<<nBlocksE, 256, 0, stream>>>(dst, node_max, ebuf, node_sum, E);
    k_scatter<<<nBlocksEW, 256, 0, stream>>>(src, dst, ebuf, node_sum, z,
                                             accum_out, E);
    k_elu<<<nBlocksNF, 256, 0, stream>>>(accum_out, N * HID);
}

extern "C" void kernel_launch(void* const* d_in, const int* in_sizes, int n_in,
                              void* d_out, int out_size, void* d_ws, size_t ws_size,
                              hipStream_t stream)
{
    // Input order (setup_inputs): h, src, dst, W1, b1, a1, ab1, W2, b2, a2, ab2
    const float* h   = (const float*)d_in[0];
    const int*   src = (const int*)  d_in[1];
    const int*   dst = (const int*)  d_in[2];
    const float* W1  = (const float*)d_in[3];
    const float* b1  = (const float*)d_in[4];
    const float* a1  = (const float*)d_in[5];
    const float* ab1 = (const float*)d_in[6];
    const float* W2  = (const float*)d_in[7];
    const float* b2  = (const float*)d_in[8];
    const float* a2  = (const float*)d_in[9];
    const float* ab2 = (const float*)d_in[10];

    const int N = in_sizes[0] / IN_DIM;   // 50000
    const int E = in_sizes[1];            // 800000

    // Workspace carve-up (all float, 16B-aligned slices)
    float* ws    = (float*)d_ws;
    float* z     = ws;                       // N*128
    float* h1    = z   + (size_t)N * HID;    // N*128
    float* ssrc  = h1  + (size_t)N * HID;    // N
    float* sdst  = ssrc + N;                 // N
    float* nmax  = sdst + N;                 // N
    float* nsum  = nmax + N;                 // N
    float* ebuf  = nsum + N;                 // E

    // Layer 1: h [N,256] -> h1 [N,128]
    run_gat_layer(h, IN_DIM, src, dst, W1, b1, a1, ab1,
                  z, ssrc, sdst, nmax, nsum, ebuf, h1, N, E, stream);

    // Layer 2: h1 [N,128] -> d_out [N,128]
    run_gat_layer(h1, HID, src, dst, W2, b2, a2, ab2,
                  z, ssrc, sdst, nmax, nsum, ebuf, (float*)d_out, N, E, stream);
}
